// RTM_89129161327203
// MI455X (gfx1250) — matrix operand order repaired and checked by reading, not compiled, not run on hardware
//
#include <hip/hip_runtime.h>
#include <hip/hip_bf16.h>

// Problem constants (from reference)
#define BATCH 2048
#define RCNT  50      // reviews per sample (padded to 64 rows for WMMA)
#define TDIM  512     // topics (K of GEMM1)
#define DDIM  64      // embed dim
#define ADIM  64      // attention dim

// LDS strides (padded to spread banks; byte offsets stay 16B-aligned)
#define RFS 68   // review f32 stride (floats)
#define RHS 72   // review bf16 stride (elements; 144B rows -> 16B aligned chunks)
#define SBS 68   // s-buffer f32 stride

typedef __attribute__((ext_vector_type(16))) __bf16 v16bf;
typedef __attribute__((ext_vector_type(8)))  __bf16 v8bf;
typedef __attribute__((ext_vector_type(8)))  float  v8f;
typedef __attribute__((ext_vector_type(4)))  float  v4f;

static __device__ inline v16bf combine16(v8bf lo, v8bf hi) {
  v16bf r;
#pragma unroll
  for (int i = 0; i < 8; ++i) { r[i] = lo[i]; r[8 + i] = hi[i]; }
  return r;
}

static __device__ inline v8f wmma_bf16(v16bf a, v16bf b, v8f c) {
  // D = A(16x32 bf16) x B(32x16 bf16) + C(16x16 f32)
  return __builtin_amdgcn_wmma_f32_16x16x32_bf16(false, a, false, b, (short)0, c,
                                                 false, false);
}

// ---------------------------------------------------------------------------
// Prep: topicT[d][k] = bf16(topic_w[k][d]); fc{ru,ri}T[j][d] = bf16(fc_*_w[d][j]);
//       colsum[d] = sum_k topic_w[k][d]
// ---------------------------------------------------------------------------
__global__ void prep_kernel(const float* __restrict__ topic_w,
                            const float* __restrict__ fc_ru_w,
                            const float* __restrict__ fc_ri_w,
                            __bf16* __restrict__ topicT,
                            __bf16* __restrict__ fcruT,
                            __bf16* __restrict__ fcriT,
                            float* __restrict__ colsum) {
  int i = blockIdx.x * blockDim.x + threadIdx.x;
  if (i < TDIM * DDIM) {                       // transpose topic table
    int d = i >> 9, k = i & 511;               // i = d*512 + k
    topicT[i] = (__bf16)topic_w[k * DDIM + d];
  } else if (i < TDIM * DDIM + DDIM * ADIM) {
    int t = i - TDIM * DDIM; int j = t >> 6, d = t & 63;
    fcruT[t] = (__bf16)fc_ru_w[d * ADIM + j];
  } else if (i < TDIM * DDIM + 2 * DDIM * ADIM) {
    int t = i - TDIM * DDIM - DDIM * ADIM; int j = t >> 6, d = t & 63;
    fcriT[t] = (__bf16)fc_ri_w[d * ADIM + j];
  } else if (i < TDIM * DDIM + 2 * DDIM * ADIM + DDIM) {
    int d = i - (TDIM * DDIM + 2 * DDIM * ADIM);
    float s = 0.f;
    for (int k = 0; k < TDIM; ++k) s += topic_w[k * DDIM + d];
    colsum[d] = s;
  }
}

// ---------------------------------------------------------------------------
// Per-(batch, branch) kernel: 256 threads = 8 waves. Two WMMA GEMMs +
// attention softmax + pooling. Writes tmp[b][branch*64 .. +63].
// ---------------------------------------------------------------------------
__global__ __launch_bounds__(256)
void branch_kernel(const int* __restrict__ user, const int* __restrict__ item,
                   const float* __restrict__ u_rt, const float* __restrict__ i_rt,
                   const float* __restrict__ u_emb, const float* __restrict__ i_emb,
                   const float* __restrict__ u_att, const float* __restrict__ i_att,
                   const float* __restrict__ fc_u_w, const float* __restrict__ fc_u_b,
                   const float* __restrict__ fc_ru_b,
                   const float* __restrict__ fc_i_w, const float* __restrict__ fc_i_b,
                   const float* __restrict__ fc_ri_b,
                   const float* __restrict__ h_u_w, const float* __restrict__ h_u_b,
                   const float* __restrict__ h_i_w, const float* __restrict__ h_i_b,
                   const __bf16* __restrict__ topicT,
                   const __bf16* __restrict__ fcruT,
                   const __bf16* __restrict__ fcriT,
                   const float* __restrict__ colsum,
                   float* __restrict__ tmp) {
  const int b    = blockIdx.x;
  const int br   = blockIdx.y;           // 0 = user branch, 1 = item branch
  const int tid  = threadIdx.x;
  const int lane = tid & 31;
  const int w    = tid >> 5;

  const int*    ids   = br ? item   : user;
  const float*  rt    = br ? i_rt   : u_rt;
  const float*  emb_w = br ? i_emb  : u_emb;
  const float*  att_w = br ? i_att  : u_att;
  const float*  fca_w = br ? fc_i_w : fc_u_w;
  const float*  fca_b = br ? fc_i_b : fc_u_b;
  const float*  fcr_b = br ? fc_ri_b : fc_ru_b;
  const __bf16* fcrT  = br ? fcriT  : fcruT;
  const float*  h_w   = br ? h_i_w  : h_u_w;
  const float*  h_b   = br ? h_i_b  : h_u_b;

  __shared__ float  reviewF[64 * RFS];   // review tile, f32 (pooling)
  __shared__ __bf16 reviewH[64 * RHS];   // review tile, bf16 (A of GEMM2)
  __shared__ float  sbuf[64 * SBS];      // relu'd attention pre-activations
  __shared__ float  afc[64];
  __shared__ float  red[64];
  __shared__ float  wgt[64];
  __shared__ float  sc[64];

  const int id = ids[b];

  // afc[j] = (att @ fc_a_w)[j] + fc_a_b[j] + fc_r_b[j]
  if (tid < ADIM) {
    float s = fca_b[tid] + fcr_b[tid];
    for (int d = 0; d < DDIM; ++d)
      s += att_w[(size_t)id * DDIM + d] * fca_w[d * ADIM + tid];
    afc[tid] = s;
  }

  // -------- GEMM1: review[64x64] = rt[b](50x512, zero-padded) x topic(512x64)
  // Tile assignment: wave w -> M-tile (w>>1), N-tiles {2*(w&1), 2*(w&1)+1}
  const int   mT    = w >> 1;
  const int   nT0   = (w & 1) * 2;
  const int   rowA  = mT * 16 + (lane & 15);
  const int   kselA = (lane >> 4) * 16;  // A half: K {0..15} vs {16..31}
  const int   kselB = (lane >> 4) * 16;  // B half: K {0..15} vs {16..31}
  const float amask = (rowA < RCNT) ? 1.0f : 0.0f;   // zero the padded rows
  const int   rowAc = (rowA < RCNT) ? rowA : 0;      // clamp to stay in bounds

  const float*  abase  = rt + ((size_t)b * RCNT + rowAc) * TDIM + kselA;
  const __bf16* b0base = topicT + (size_t)(nT0 * 16 + (lane & 15)) * TDIM + kselB;
  const __bf16* b1base = b0base + (size_t)16 * TDIM;

  v8f c0 = {}; v8f c1 = {};
#pragma unroll 4
  for (int ks = 0; ks < TDIM / 32; ++ks) {
    const float* ap = abase + ks * 32;
    // Single-use 420MB stream: non-temporal so it does not churn the 192MB L2
    // (keeps topicT / fc tables / embedding rows resident at RT priority).
    v4f f0 = __builtin_nontemporal_load((const v4f*)(ap));
    v4f f1 = __builtin_nontemporal_load((const v4f*)(ap + 4));
    v4f f2 = __builtin_nontemporal_load((const v4f*)(ap + 8));
    v4f f3 = __builtin_nontemporal_load((const v4f*)(ap + 12));
    v16bf a;
#pragma unroll
    for (int i = 0; i < 4; ++i) {
      a[i]      = (__bf16)(f0[i] * amask);
      a[4 + i]  = (__bf16)(f1[i] * amask);
      a[8 + i]  = (__bf16)(f2[i] * amask);
      a[12 + i] = (__bf16)(f3[i] * amask);
    }
    v16bf bb0 = combine16(*(const v8bf*)(b0base + ks * 32),
                          *(const v8bf*)(b0base + ks * 32 + 8));
    v16bf bb1 = combine16(*(const v8bf*)(b1base + ks * 32),
                          *(const v8bf*)(b1base + ks * 32 + 8));
    c0 = wmma_bf16(a, bb0, c0);
    c1 = wmma_bf16(a, bb1, c1);
  }

  // Scatter C: lanes 0-15 rows v, lanes 16-31 rows v+8; col = nTile*16+(lane&15)
  {
    const int col0  = nT0 * 16 + (lane & 15);
    const int col1  = col0 + 16;
    const int rbase = mT * 16 + ((lane >> 4) * 8);
#pragma unroll
    for (int v = 0; v < 8; ++v) {
      const int row = rbase + v;
      reviewF[row * RFS + col0] = c0[v];
      reviewF[row * RFS + col1] = c1[v];
      reviewH[row * RHS + col0] = (__bf16)c0[v];
      reviewH[row * RHS + col1] = (__bf16)c1[v];
    }
  }
  __syncthreads();

  // -------- GEMM2: s_pre[64x64] = review(bf16) x fc_r_w(64x64), K = 64
  v8f d0 = {}; v8f d1 = {};
  {
    const __bf16* arow  = &reviewH[(size_t)(mT * 16 + (lane & 15)) * RHS];
    const __bf16* brow0 = fcrT + (size_t)(nT0 * 16 + (lane & 15)) * DDIM + kselB;
    const __bf16* brow1 = brow0 + (size_t)16 * DDIM;
#pragma unroll
    for (int ks = 0; ks < 2; ++ks) {
      const int kb = ks * 32 + kselA;
      v16bf a   = combine16(*(const v8bf*)(arow + kb),
                            *(const v8bf*)(arow + kb + 8));
      v16bf bb0 = combine16(*(const v8bf*)(brow0 + ks * 32),
                            *(const v8bf*)(brow0 + ks * 32 + 8));
      v16bf bb1 = combine16(*(const v8bf*)(brow1 + ks * 32),
                            *(const v8bf*)(brow1 + ks * 32 + 8));
      d0 = wmma_bf16(a, bb0, d0);
      d1 = wmma_bf16(a, bb1, d1);
    }
  }

  // s = relu(afc[col] + s_pre)
  {
    const int   col0  = nT0 * 16 + (lane & 15);
    const int   col1  = col0 + 16;
    const float base0 = afc[col0];
    const float base1 = afc[col1];
    const int   rbase = mT * 16 + ((lane >> 4) * 8);
#pragma unroll
    for (int v = 0; v < 8; ++v) {
      float s0 = base0 + d0[v]; s0 = s0 > 0.f ? s0 : 0.f;
      float s1 = base1 + d1[v]; s1 = s1 > 0.f ? s1 : 0.f;
      sbuf[(rbase + v) * SBS + col0] = s0;
      sbuf[(rbase + v) * SBS + col1] = s1;
    }
  }
  __syncthreads();

  // Attention scores: score[r] = s[r] . h_w + h_b
  if (tid < 64) {
    float s = -3.0e38f;
    if (tid < RCNT) {
      float acc = h_b[0];
      for (int j = 0; j < ADIM; ++j) acc += sbuf[tid * SBS + j] * h_w[j];
      sc[tid] = acc;
      s = acc;
    }
    red[tid] = s;
  }
  __syncthreads();
  for (int off = 32; off > 0; off >>= 1) {       // max reduce
    if (tid < off) red[tid] = fmaxf(red[tid], red[tid + off]);
    __syncthreads();
  }
  const float mx = red[0];
  __syncthreads();
  if (tid < 64) {
    float e = (tid < RCNT) ? __expf(sc[tid] - mx) : 0.f;
    wgt[tid] = e;
    red[tid] = e;
  }
  __syncthreads();
  for (int off = 32; off > 0; off >>= 1) {       // sum reduce
    if (tid < off) red[tid] += red[tid + off];
    __syncthreads();
  }
  const float denom = red[0];
  __syncthreads();

  // Pooling: feat[d] = (sum_r w[r]*review[r][d]) * colsum[d] + emb[d]
  if (tid < DDIM) {
    float f = 0.f;
    for (int r = 0; r < RCNT; ++r) f += wgt[r] * reviewF[r * RFS + tid];
    f = (f / denom) * colsum[tid] + emb_w[(size_t)id * DDIM + tid];
    tmp[(size_t)b * 128 + br * 64 + tid] = f;
  }
}

// ---------------------------------------------------------------------------
// Final projection: pred[b] = relu(tmp[b] . fc_pre_w + fc_pre_b)
// ---------------------------------------------------------------------------
__global__ __launch_bounds__(128)
void pred_kernel(const float* __restrict__ tmp,
                 const float* __restrict__ fc_pre_w,
                 const float* __restrict__ fc_pre_b,
                 float* __restrict__ out) {
  __shared__ float r[128];
  const int b = blockIdx.x;
  const int t = threadIdx.x;
  r[t] = tmp[(size_t)b * 128 + t] * fc_pre_w[t];
  __syncthreads();
  for (int off = 64; off > 0; off >>= 1) {
    if (t < off) r[t] += r[t + off];
    __syncthreads();
  }
  if (t == 0) {
    float v = r[0] + fc_pre_b[0];
    out[b] = v > 0.f ? v : 0.f;
  }
}

// ---------------------------------------------------------------------------
extern "C" void kernel_launch(void* const* d_in, const int* in_sizes, int n_in,
                              void* d_out, int out_size, void* d_ws, size_t ws_size,
                              hipStream_t stream) {
  (void)in_sizes; (void)n_in; (void)out_size; (void)ws_size;
  const int*   user     = (const int*)d_in[0];
  const int*   item     = (const int*)d_in[1];
  const float* u_rt     = (const float*)d_in[2];
  const float* i_rt     = (const float*)d_in[3];
  const float* u_emb    = (const float*)d_in[4];
  const float* i_emb    = (const float*)d_in[5];
  const float* u_att    = (const float*)d_in[6];
  const float* i_att    = (const float*)d_in[7];
  const float* topic    = (const float*)d_in[8];
  const float* fc_u_w   = (const float*)d_in[9];
  const float* fc_u_b   = (const float*)d_in[10];
  const float* fc_ru_w  = (const float*)d_in[11];
  const float* fc_ru_b  = (const float*)d_in[12];
  const float* fc_i_w   = (const float*)d_in[13];
  const float* fc_i_b   = (const float*)d_in[14];
  const float* fc_ri_w  = (const float*)d_in[15];
  const float* fc_ri_b  = (const float*)d_in[16];
  const float* h_u_w    = (const float*)d_in[17];
  const float* h_u_b    = (const float*)d_in[18];
  const float* h_i_w    = (const float*)d_in[19];
  const float* h_i_b    = (const float*)d_in[20];
  const float* fc_pre_w = (const float*)d_in[21];
  const float* fc_pre_b = (const float*)d_in[22];

  // Workspace layout (bytes):
  //   [0,      65536)  topicT   bf16 [64][512]
  //   [65536,  73728)  fcruT    bf16 [64][64]
  //   [73728,  81920)  fcriT    bf16 [64][64]
  //   [81920,  82176)  colsum   f32  [64]
  //   [82176, +1MB  )  tmp      f32  [2048][128]
  char* ws = (char*)d_ws;
  __bf16* topicT = (__bf16*)(ws + 0);
  __bf16* fcruT  = (__bf16*)(ws + 65536);
  __bf16* fcriT  = (__bf16*)(ws + 73728);
  float*  colsum = (float*)(ws + 81920);
  float*  tmp    = (float*)(ws + 82176);
  float*  out    = (float*)d_out;

  const int prep_elems = TDIM * DDIM + 2 * DDIM * ADIM + DDIM;  // 41024
  prep_kernel<<<dim3((prep_elems + 255) / 256), dim3(256), 0, stream>>>(
      topic, fc_ru_w, fc_ri_w, topicT, fcruT, fcriT, colsum);

  branch_kernel<<<dim3(BATCH, 2), dim3(256), 0, stream>>>(
      user, item, u_rt, i_rt, u_emb, i_emb, u_att, i_att,
      fc_u_w, fc_u_b, fc_ru_b, fc_i_w, fc_i_b, fc_ri_b,
      h_u_w, h_u_b, h_i_w, h_i_b,
      topicT, fcruT, fcriT, colsum, tmp);

  pred_kernel<<<dim3(BATCH), dim3(128), 0, stream>>>(tmp, fc_pre_w, fc_pre_b, out);
}
